// DTSGNet_90082644066744
// MI455X (gfx1250) — matrix operand order repaired and checked by reading, not compiled, not run on hardware
//
#include <hip/hip_runtime.h>
#include <hip/hip_bf16.h>
#include <stdint.h>

typedef __attribute__((ext_vector_type(16))) __bf16 v16bf;
typedef __attribute__((ext_vector_type(8)))  float  v8f;
typedef unsigned short u16;
typedef unsigned int   u32;

#define B_SZ   1024
#define L_SZ   1024
#define LP     64
#define STRIDE 32
#define WIN    8
#define GH     64
#define RH     128
#define NC     11
#define P_CNT  31    // (1024-64)/32 + 1
#define T_CNT  31

// ---------------- helpers ----------------

// Hardware f32 -> bf16 conversion (RTNE), lowers to v_cvt(_pk)_bf16_f32.
__device__ __forceinline__ u16 f2bf(float x) {
    __bf16 h = (__bf16)x;
    return __builtin_bit_cast(u16, h);
}
__device__ __forceinline__ float bf2f(u16 h) {
    u32 u = ((u32)h) << 16;
    return __builtin_bit_cast(float, u);
}
__device__ __forceinline__ float sigm(float x) {
    return 1.0f / (1.0f + __expf(-x));
}

union FragBF { u32 u[8]; v16bf v; };

// A-fragment: 16x32 bf16 tile from row-major bf16 matrix (ld = row stride, elems).
// Lanes 0-15: rows M=0..15, K pairs {0..7,16..23}; lanes 16-31: same rows, K {8..15,24..31}.
__device__ __forceinline__ v16bf load_fragA(const u16* base, int row0, int ld, int kc) {
    FragBF f;
    int lane = threadIdx.x & 31;
    int half = lane >> 4, r = lane & 15;
    const u16* p = base + (size_t)(row0 + r) * (size_t)ld;
    int kb = kc * 32 + half * 8;
#pragma unroll
    for (int j = 0; j < 8; ++j) {
        int k = kb + ((j >> 2) << 4) + ((j & 3) << 1);
        f.u[j] = *(const u32*)(p + k);
    }
    return f.v;
}

// B-fragment: 32x16 bf16 tile from pair-packed [K/2][N] u32 matrix (ldn = N).
// word(kk,n) = bf16(B[2kk][n]) | bf16(B[2kk+1][n])<<16.
__device__ __forceinline__ v16bf load_fragB(const u32* base, int kc, int col0, int ldn) {
    FragBF f;
    int lane = threadIdx.x & 31;
    int half = lane >> 4, r = lane & 15;
    const u32* p = base + (size_t)(kc * 16 + half * 4) * (size_t)ldn + col0 + r;
#pragma unroll
    for (int j = 0; j < 8; ++j) f.u[j] = p[(size_t)(((j >> 2) << 3) + (j & 3)) * ldn];
    return f.v;
}

__device__ __forceinline__ v8f wmma_bf16(v16bf a, v16bf b, v8f c) {
    return __builtin_amdgcn_wmma_f32_16x16x32_bf16(false, a, false, b, (short)0, c, false, false);
}

// ---------------- setup kernels ----------------

// Row-normalized banded adjacency -> bf16 row-major [64][64]
__global__ void setup_adj_kernel(const float* __restrict__ ew, u16* __restrict__ adjA) {
    int i = threadIdx.x;           // 0..63, one row per thread
    if (i >= LP) return;
    float row[LP];
    float deg = 0.f;
#pragma unroll 8
    for (int j = 0; j < LP; ++j) {
        int d = i - j; if (d < 0) d = -d;
        float m = (d <= WIN && d != 0) ? 1.f : 0.f;
        float a = sigm(ew[i * LP + j]) * m;
        row[j] = a; deg += a;
    }
    float inv = deg > 0.f ? 1.f / deg : 0.f;
#pragma unroll 8
    for (int j = 0; j < LP; ++j) adjA[i * LP + j] = f2bf(row[j] * inv);
}

// Pack W[O][K] (row-major f32) into WMMA B layout: dst[kk*O + o] = pair(W[o][2kk],W[o][2kk+1])
__global__ void pack_bmat_kernel(const float* __restrict__ W, u32* __restrict__ dst,
                                 int O, int K) {
    int idx = blockIdx.x * 256 + threadIdx.x;
    int total = (K >> 1) * O;
    if (idx >= total) return;
    int kk = idx / O, o = idx - kk * O;
    u32 lo = f2bf(W[(size_t)o * K + 2 * kk]);
    u32 hi = f2bf(W[(size_t)o * K + 2 * kk + 1]);
    dst[idx] = lo | (hi << 16);
}

__global__ void vecadd_kernel(const float* __restrict__ a, const float* __restrict__ b,
                              float* __restrict__ c, int n) {
    int i = blockIdx.x * 256 + threadIdx.x;
    if (i < n) c[i] = a[i] + b[i];
}

// ---------------- fused SAGE1 + SAGE2 + node-sum ----------------
// grid = B*P blocks, 256 threads (8 wave32). Output: feats_bf [B][P][GH] bf16.

__global__ __launch_bounds__(256) void sage_kernel(
    const float* __restrict__ I, const float* __restrict__ Q,
    const float* __restrict__ Wn1, const float* __restrict__ bn1,
    const float* __restrict__ Ws1, const float* __restrict__ bs1,
    const u16* __restrict__ adjA_g, const u32* __restrict__ Ws2B_g,
    const u32* __restrict__ Wn2B_g, const float* __restrict__ bsum2,
    u16* __restrict__ feats_bf)
{
    __shared__ u16 adjA[LP * LP];      // A' bf16 row-major (A-role)
    __shared__ u16 h1A[LP * GH];       // h1 row-major (A-role)
    __shared__ u32 h1B[(LP / 2) * GH]; // h1 pair-packed over node dim (B-role)
    __shared__ u16 n2A[LP * GH];       // neigh2 row-major (A-role)
    __shared__ u32 WsB[(GH / 2) * GH];
    __shared__ u32 WnB[(GH / 2) * GH];
    __shared__ float x0[LP * 2];
    __shared__ float featacc[GH];

    const int t = threadIdx.x;
    const int b = blockIdx.x / P_CNT;
    const int p = blockIdx.x - b * P_CNT;

    // stage adjacency + packed SAGE2 weights into LDS
    {
        const u32* asrc = (const u32*)adjA_g;
        u32* adst = (u32*)adjA;
#pragma unroll
        for (int i = t; i < 2048; i += 256) {
            adst[i] = asrc[i];
            WsB[i]  = Ws2B_g[i];
            WnB[i]  = Wn2B_g[i];
        }
    }
    if (t < LP) {
        x0[t * 2 + 0] = I[(size_t)b * L_SZ + p * STRIDE + t];
        x0[t * 2 + 1] = Q[(size_t)b * L_SZ + p * STRIDE + t];
        featacc[t] = 0.f;
    }
    __syncthreads();

    // ---- SAGE layer 1 (d=2 -> 64), plain VALU; thread -> (node, 16 outputs)
    {
        int node = t >> 2, ob = (t & 3) << 4;
        float xa = x0[node * 2], xb = x0[node * 2 + 1];
        float na = 0.f, nb = 0.f;
        int j0 = node - WIN < 0 ? 0 : node - WIN;
        int j1 = node + WIN > LP - 1 ? LP - 1 : node + WIN;
        for (int j = j0; j <= j1; ++j) {
            float a = bf2f(adjA[node * LP + j]);   // diagonal already 0
            na += a * x0[j * 2];
            nb += a * x0[j * 2 + 1];
        }
#pragma unroll
        for (int o = ob; o < ob + 16; ++o) {
            float v = Ws1[o * 2] * xa + Ws1[o * 2 + 1] * xb + bs1[o]
                    + Wn1[o * 2] * na + Wn1[o * 2 + 1] * nb + bn1[o];
            v = v > 0.f ? v : 0.f;
            h1A[node * GH + o] = f2bf(v);
        }
    }
    __syncthreads();

    // pair-pack h1 over node dim for its B-role in adj@h1
    for (int w = t; w < (LP / 2) * GH; w += 256) {
        int kk = w >> 6, n = w & 63;
        h1B[w] = (u32)h1A[(2 * kk) * GH + n] | ((u32)h1A[(2 * kk + 1) * GH + n] << 16);
    }
    __syncthreads();

    const int wave = t >> 5;
    const int lane = t & 31;
    const int half = lane >> 4, r = lane & 15;

    // ---- GEMM1: neigh2 = A' @ h1  (64x64x64, WMMA)
#pragma unroll
    for (int tt = 0; tt < 2; ++tt) {
        int tile = wave + tt * 8;          // 0..15
        int mi = tile >> 2, ni = tile & 3;
        v8f c = {};
#pragma unroll
        for (int kc = 0; kc < 2; ++kc) {
            v16bf a  = load_fragA(adjA, mi * 16, LP, kc);
            v16bf bb = load_fragB(h1B, kc, ni * 16, GH);
            c = wmma_bf16(a, bb, c);
        }
#pragma unroll
        for (int vr = 0; vr < 8; ++vr)
            n2A[(mi * 16 + vr + half * 8) * GH + ni * 16 + r] = f2bf(c[vr]);
    }
    __syncthreads();

    // ---- GEMM2+3: out2 = relu(h1@Ws2^T + neigh2@Wn2^T + bias); column-sum to feats
#pragma unroll
    for (int tt = 0; tt < 2; ++tt) {
        int tile = wave + tt * 8;
        int mi = tile >> 2, ni = tile & 3;
        v8f c = {};
#pragma unroll
        for (int kc = 0; kc < 2; ++kc)
            c = wmma_bf16(load_fragA(h1A, mi * 16, GH, kc),
                          load_fragB(WsB, kc, ni * 16, GH), c);
#pragma unroll
        for (int kc = 0; kc < 2; ++kc)
            c = wmma_bf16(load_fragA(n2A, mi * 16, GH, kc),
                          load_fragB(WnB, kc, ni * 16, GH), c);
        float bias = bsum2[ni * 16 + r];
        float s = 0.f;
#pragma unroll
        for (int vr = 0; vr < 8; ++vr) {
            float v = c[vr] + bias;
            v = v > 0.f ? v : 0.f;
            s += v;
        }
        atomicAdd(&featacc[ni * 16 + r], s);   // ds_add_f32
    }
    __syncthreads();

    if (t < GH) feats_bf[((size_t)b * P_CNT + p) * GH + t] = f2bf(featacc[t]);
}

// ---------------- LSTM layer (optionally + classifier) ----------------
// grid = B/16 blocks, 256 threads (8 wave32). Each block: 16 batch rows, all T steps.
// Recurrent + input GEMMs via WMMA; weights read pair-packed from scratch (L2-hot).

__global__ __launch_bounds__(256) void lstm_kernel(
    const u16* __restrict__ xin, int K,          // [B][T][K] bf16 row-major
    const u32* __restrict__ WiB,                 // [K/2][4RH] packed
    const u32* __restrict__ WhB,                 // [RH/2][4RH] packed
    const float* __restrict__ bias,              // [4RH] = bih + bhh
    u16* __restrict__ outseq,                    // [B][T][RH] bf16 or nullptr
    const float* __restrict__ Wc1, const float* __restrict__ bc1,
    const float* __restrict__ Wc2, const float* __restrict__ bc2,
    float* __restrict__ out)                     // [B][NC] or nullptr
{
    __shared__ u16   hA[16 * RH];        // h (bf16) for A-role
    __shared__ float cst[16 * RH];       // cell state
    __shared__ float hF[16 * RH];        // h (f32) for classifier
    __shared__ float gpre[16 * 4 * RH];  // gate pre-activations
    __shared__ float hid[16 * 64];

    const int t = threadIdx.x;
    const int b0 = blockIdx.x * 16;
    const int wave = t >> 5, lane = t & 31;
    const int half = lane >> 4, r = lane & 15;
    const int kcI = K >> 5;              // input K chunks (2 or 4)

    for (int i = t; i < 16 * RH; i += 256) { hA[i] = 0; cst[i] = 0.f; }
    __syncthreads();

    for (int step = 0; step < T_CNT; ++step) {
        // A-frags: x_t from global (row = batch b0+r, ld = T*K elems), h from LDS
        v16bf ax[4], ah[4];
        for (int kc = 0; kc < kcI; ++kc)
            ax[kc] = load_fragA(xin + (size_t)step * K, b0, T_CNT * K, kc);
#pragma unroll
        for (int kc = 0; kc < 4; ++kc)
            ah[kc] = load_fragA(hA, 0, RH, kc);

#pragma unroll
        for (int q = 0; q < 4; ++q) {
            int ncol = (wave * 4 + q) * 16;
            v8f c = {};
            for (int kc = 0; kc < kcI; ++kc)
                c = wmma_bf16(ax[kc], load_fragB(WiB, kc, ncol, 4 * RH), c);
#pragma unroll
            for (int kc = 0; kc < 4; ++kc)
                c = wmma_bf16(ah[kc], load_fragB(WhB, kc, ncol, 4 * RH), c);
#pragma unroll
            for (int vr = 0; vr < 8; ++vr)
                gpre[(vr + half * 8) * (4 * RH) + ncol + r] = c[vr];
        }
        __syncthreads();

        // elementwise gates (PyTorch order i,f,g,o)
        for (int e = t; e < 16 * RH; e += 256) {
            int row = e >> 7, j = e & (RH - 1);
            const float* g = &gpre[row * 4 * RH];
            float gi = g[j]           + bias[j];
            float gf = g[RH + j]      + bias[RH + j];
            float gg = g[2 * RH + j]  + bias[2 * RH + j];
            float go = g[3 * RH + j]  + bias[3 * RH + j];
            float cn = sigm(gf) * cst[e] + sigm(gi) * tanhf(gg);
            float hn = sigm(go) * tanhf(cn);
            cst[e] = cn;
            hF[e]  = hn;
            hA[e]  = f2bf(hn);
            if (outseq)
                outseq[((size_t)(b0 + row) * T_CNT + step) * RH + j] = f2bf(hn);
        }
        __syncthreads();
    }

    if (Wc1) {
        // hid = relu(h @ Wc1^T + bc1)  [16,64]
        for (int e = t; e < 16 * 64; e += 256) {
            int row = e >> 6, o = e & 63;
            float s = bc1[o];
            for (int d = 0; d < RH; ++d) s += hF[row * RH + d] * Wc1[o * RH + d];
            hid[e] = s > 0.f ? s : 0.f;
        }
        __syncthreads();
        // out = hid @ Wc2^T + bc2  [16,NC]
        for (int e = t; e < 16 * NC; e += 256) {
            int row = e / NC, cix = e - row * NC;
            float s = bc2[cix];
            for (int d = 0; d < 64; ++d) s += hid[row * 64 + d] * Wc2[cix * 64 + d];
            out[(size_t)(b0 + row) * NC + cix] = s;
        }
    }
}

// ---------------- host launcher ----------------

extern "C" void kernel_launch(void* const* d_in, const int* in_sizes, int n_in,
                              void* d_out, int out_size, void* d_ws, size_t ws_size,
                              hipStream_t stream) {
    (void)in_sizes; (void)n_in; (void)out_size; (void)ws_size;

    const float* I    = (const float*)d_in[0];
    const float* Q    = (const float*)d_in[1];
    const float* ew   = (const float*)d_in[2];
    const float* Wn1  = (const float*)d_in[3];
    const float* bn1  = (const float*)d_in[4];
    const float* Ws1  = (const float*)d_in[5];
    const float* bs1  = (const float*)d_in[6];
    const float* Wn2  = (const float*)d_in[7];
    const float* bn2  = (const float*)d_in[8];
    const float* Ws2  = (const float*)d_in[9];
    const float* bs2  = (const float*)d_in[10];
    const float* Wih0 = (const float*)d_in[11];
    const float* Whh0 = (const float*)d_in[12];
    const float* bih0 = (const float*)d_in[13];
    const float* bhh0 = (const float*)d_in[14];
    const float* Wih1 = (const float*)d_in[15];
    const float* Whh1 = (const float*)d_in[16];
    const float* bih1 = (const float*)d_in[17];
    const float* bhh1 = (const float*)d_in[18];
    const float* Wc1  = (const float*)d_in[19];
    const float* bc1  = (const float*)d_in[20];
    const float* Wc2  = (const float*)d_in[21];
    const float* bc2  = (const float*)d_in[22];
    float* outp = (float*)d_out;

    // scratch carve-up (256B aligned)
    char* ws = (char*)d_ws;
    size_t off = 0;
    auto take = [&](size_t bytes) {
        void* p = ws + off;
        off = (off + bytes + 255) & ~(size_t)255;
        return p;
    };
    u16* adjA   = (u16*)take(LP * LP * 2);
    u32* Ws2B   = (u32*)take((GH / 2) * GH * 4);
    u32* Wn2B   = (u32*)take((GH / 2) * GH * 4);
    u32* Wih0B  = (u32*)take((GH / 2) * 4 * RH * 4);
    u32* Whh0B  = (u32*)take((RH / 2) * 4 * RH * 4);
    u32* Wih1B  = (u32*)take((RH / 2) * 4 * RH * 4);
    u32* Whh1B  = (u32*)take((RH / 2) * 4 * RH * 4);
    float* bsum2 = (float*)take(GH * 4);
    float* bihh0 = (float*)take(4 * RH * 4);
    float* bihh1 = (float*)take(4 * RH * 4);
    u16* feats  = (u16*)take((size_t)B_SZ * P_CNT * GH * 2);
    u16* out0   = (u16*)take((size_t)B_SZ * T_CNT * RH * 2);

    // ---- setup: adjacency + weight packing + bias sums
    setup_adj_kernel<<<1, 64, 0, stream>>>(ew, adjA);
    pack_bmat_kernel<<<((GH / 2) * GH + 255) / 256, 256, 0, stream>>>(Ws2, Ws2B, GH, GH);
    pack_bmat_kernel<<<((GH / 2) * GH + 255) / 256, 256, 0, stream>>>(Wn2, Wn2B, GH, GH);
    pack_bmat_kernel<<<((GH / 2) * 4 * RH + 255) / 256, 256, 0, stream>>>(Wih0, Wih0B, 4 * RH, GH);
    pack_bmat_kernel<<<((RH / 2) * 4 * RH + 255) / 256, 256, 0, stream>>>(Whh0, Whh0B, 4 * RH, RH);
    pack_bmat_kernel<<<((RH / 2) * 4 * RH + 255) / 256, 256, 0, stream>>>(Wih1, Wih1B, 4 * RH, RH);
    pack_bmat_kernel<<<((RH / 2) * 4 * RH + 255) / 256, 256, 0, stream>>>(Whh1, Whh1B, 4 * RH, RH);
    vecadd_kernel<<<1, 256, 0, stream>>>(bs2, bn2, bsum2, GH);
    vecadd_kernel<<<2, 256, 0, stream>>>(bih0, bhh0, bihh0, 4 * RH);
    vecadd_kernel<<<2, 256, 0, stream>>>(bih1, bhh1, bihh1, 4 * RH);

    // ---- fused SAGE (hotspot): one block per (batch, patch)
    sage_kernel<<<B_SZ * P_CNT, 256, 0, stream>>>(
        I, Q, Wn1, bn1, Ws1, bs1, adjA, Ws2B, Wn2B, bsum2, feats);

    // ---- LSTM layer 0 (emit full sequence)
    lstm_kernel<<<B_SZ / 16, 256, 0, stream>>>(
        feats, GH, Wih0B, Whh0B, bihh0, out0,
        nullptr, nullptr, nullptr, nullptr, nullptr);

    // ---- LSTM layer 1 + classifier (final hidden only)
    lstm_kernel<<<B_SZ / 16, 256, 0, stream>>>(
        out0, RH, Wih1B, Whh1B, bihh1, nullptr,
        Wc1, bc1, Wc2, bc2, outp);
}
